// Calculate_Attention_40389872452067
// MI455X (gfx1250) — compile-verified, matrix operand order checked by reading, not run on hardware
//
#include <hip/hip_runtime.h>

typedef __attribute__((ext_vector_type(16))) _Float16 v16h;
typedef __attribute__((ext_vector_type(8)))  _Float16 v8h;
typedef __attribute__((ext_vector_type(8)))  float    v8f;

#define NHEADS 96      // B*H = 8*12
#define SEQ    1024
#define DIM    64
#define KVSTEP 32
#define NSTEP  (SEQ / KVSTEP)

// (1/sqrt(64)) * log2(e): folded into the Q fragments so the whole softmax
// runs in the log2 domain (exp -> bare v_exp_f32, no per-step scaling).
#define QSCALE 0.18033688011112042f

__device__ __forceinline__ v16h cat16(v8h lo, v8h hi) {
    return __builtin_shufflevector(lo, hi, 0, 1, 2, 3, 4, 5, 6, 7,
                                           8, 9, 10, 11, 12, 13, 14, 15);
}

__device__ __forceinline__ v8h cvt8(float4 a, float4 b) {
    v8h h;
    h[0] = (_Float16)a.x; h[1] = (_Float16)a.y; h[2] = (_Float16)a.z; h[3] = (_Float16)a.w;
    h[4] = (_Float16)b.x; h[5] = (_Float16)b.y; h[6] = (_Float16)b.z; h[7] = (_Float16)b.w;
    return h;
}

__device__ __forceinline__ v8h cvt8s(float4 a, float4 b, float s) {
    v8h h;
    h[0] = (_Float16)(a.x * s); h[1] = (_Float16)(a.y * s);
    h[2] = (_Float16)(a.z * s); h[3] = (_Float16)(a.w * s);
    h[4] = (_Float16)(b.x * s); h[5] = (_Float16)(b.y * s);
    h[6] = (_Float16)(b.z * s); h[7] = (_Float16)(b.w * s);
    return h;
}

__global__ __launch_bounds__(256)
void flash_attn_fwd_gfx1250(const float* __restrict__ Q,
                            const float* __restrict__ K,
                            const float* __restrict__ V,
                            float* __restrict__ O) {
    // Double-buffered f16 tiles, converted once at stage time and shared by all
    // 8 waves: K row-major (k,d); V pre-transposed (d,k) so the second-matmul
    // A-fragments are contiguous 16B runs.
    __shared__ _Float16 sK [2][KVSTEP][DIM];   // 2 x 4 KB
    __shared__ _Float16 sVT[2][DIM][KVSTEP];   // 2 x 4 KB

    const int tid  = threadIdx.x;
    const int lane = tid & 31;
    const int wave = tid >> 5;
    const int l    = lane & 15;
    const int hi   = lane >> 4;
    const int bh   = blockIdx.y;
    const int q0   = blockIdx.x * 128 + wave * 16;   // 16 query rows per wave

    const size_t head = (size_t)bh * SEQ * DIM;
    const float* Qh = Q + head;
    const float* Kh = K + head;
    const float* Vh = V + head;
    float*       Oh = O + head;

    // Staging coords: thread owns an 8-wide sliver of one tile row.
    const int sk = tid >> 3;         // 0..31 : row (k) within tile
    const int sd = (tid & 7) * 8;    // 0..56 : col (d) start

    // ---- Q^T B-fragments (persist), pre-scaled by (1/sqrt(d))*log2(e).
    v16h qb[2];
#pragma unroll
    for (int c = 0; c < 2; ++c) {
        const float4* r4 = reinterpret_cast<const float4*>(
            Qh + (size_t)(q0 + l) * DIM + c * 32 + hi * 8);
        qb[c] = cat16(cvt8s(r4[0], r4[1], QSCALE), cvt8s(r4[4], r4[5], QSCALE));
    }

    // ---- prologue: stage tile 0
    {
        const float4* kg = reinterpret_cast<const float4*>(Kh + (size_t)sk * DIM + sd);
        const float4* vg = reinterpret_cast<const float4*>(Vh + (size_t)sk * DIM + sd);
        float4 k0 = kg[0], k1 = kg[1], v0 = vg[0], v1 = vg[1];
        *reinterpret_cast<v8h*>(&sK[0][sk][sd]) = cvt8(k0, k1);
        v8h hv = cvt8(v0, v1);
#pragma unroll
        for (int j = 0; j < 8; ++j) sVT[0][sd + j][sk] = hv[j];
    }
    __syncthreads();

    // O^T accumulator: accT[ch][r] = O^T[d = ch*16 + r + 8*hi][q = q0+l]
    v8f accT[4] = {};
    float mrow = -1e30f;   // running max, log2 units
    float lrow = 0.0f;     // running sum, linear units

    for (int step = 0; step < NSTEP; ++step) {
        const int  cur  = step & 1;
        const bool more = (step + 1 < NSTEP);

        // ---- issue next tile's global loads early (latency hidden by compute)
        float4 nk0 = {}, nk1 = {}, nv0 = {}, nv1 = {};
        if (more) {
            const size_t nbase = (size_t)((step + 1) * KVSTEP + sk) * DIM + sd;
            const float4* kg = reinterpret_cast<const float4*>(Kh + nbase);
            const float4* vg = reinterpret_cast<const float4*>(Vh + nbase);
            nk0 = kg[0]; nk1 = kg[1]; nv0 = vg[0]; nv1 = vg[1];
        }

        // ---- S^T tiles (already in log2 units): A = K rows, B = scaled Q^T
        v8f s0 = {}, s1 = {};
#pragma unroll
        for (int c = 0; c < 2; ++c) {
            const _Float16* kp = &sK[cur][l][c * 32 + hi * 8];
            v16h ka = cat16(*reinterpret_cast<const v8h*>(kp),
                            *reinterpret_cast<const v8h*>(kp + 16));
            s0 = __builtin_amdgcn_wmma_f32_16x16x32_f16(false, ka, false, qb[c],
                                                        (short)0, s0, false, false);
        }
#pragma unroll
        for (int c = 0; c < 2; ++c) {
            const _Float16* kp = &sK[cur][16 + l][c * 32 + hi * 8];
            v16h ka = cat16(*reinterpret_cast<const v8h*>(kp),
                            *reinterpret_cast<const v8h*>(kp + 16));
            s1 = __builtin_amdgcn_wmma_f32_16x16x32_f16(false, ka, false, qb[c],
                                                        (short)0, s1, false, false);
        }

        // ---- online softmax (log2 domain) for q = q0+l
        float tmax = fmaxf(s0[0], s1[0]);
#pragma unroll
        for (int r = 1; r < 8; ++r) tmax = fmaxf(tmax, fmaxf(s0[r], s1[r]));
        tmax = fmaxf(tmax, __shfl_xor(tmax, 16, 32));

        float mnew = fmaxf(mrow, tmax);
        float corr = __builtin_amdgcn_exp2f(mrow - mnew);

        // P^T fragment: Sᵀ C-layout matches the B-layout lane-for-lane.
        v16h pb;
        float psum = 0.0f;
#pragma unroll
        for (int r = 0; r < 8; ++r) {
            float p0 = __builtin_amdgcn_exp2f(s0[r] - mnew);
            float p1 = __builtin_amdgcn_exp2f(s1[r] - mnew);
            psum += p0 + p1;
            pb[r]     = (_Float16)p0;
            pb[8 + r] = (_Float16)p1;
        }
        psum += __shfl_xor(psum, 16, 32);

        lrow = lrow * corr + psum;
        mrow = mnew;
#pragma unroll
        for (int ch = 0; ch < 4; ++ch)
#pragma unroll
            for (int r = 0; r < 8; ++r) accT[ch][r] *= corr;

        // ---- O^T += V^T · P^T ; A-fragments read straight from transposed tile
#pragma unroll
        for (int ch = 0; ch < 4; ++ch) {
            const _Float16* vp = &sVT[cur][ch * 16 + l][hi * 8];
            v16h va = cat16(*reinterpret_cast<const v8h*>(vp),
                            *reinterpret_cast<const v8h*>(vp + 16));
            accT[ch] = __builtin_amdgcn_wmma_f32_16x16x32_f16(false, va, false, pb,
                                                              (short)0, accT[ch], false, false);
        }

        // ---- write next tile into the other buffer (safe: last read of it
        // completed before the previous barrier), then sync.
        if (more) {
            *reinterpret_cast<v8h*>(&sK[cur ^ 1][sk][sd]) = cvt8(nk0, nk1);
            v8h hv = cvt8(nv0, nv1);
#pragma unroll
            for (int j = 0; j < 8; ++j) sVT[cur ^ 1][sd + j][sk] = hv[j];
        }
        __syncthreads();
    }

    // ---- epilogue: O[q][d] = accT / lrow ; lane (l,hi) owns row q0+l
    float inv = 1.0f / lrow;
#pragma unroll
    for (int ch = 0; ch < 4; ++ch) {
        float* orow = Oh + (size_t)(q0 + l) * DIM + ch * 16 + hi * 8;
        float4* o4 = reinterpret_cast<float4*>(orow);
        o4[0] = make_float4(accT[ch][0] * inv, accT[ch][1] * inv,
                            accT[ch][2] * inv, accT[ch][3] * inv);
        o4[1] = make_float4(accT[ch][4] * inv, accT[ch][5] * inv,
                            accT[ch][6] * inv, accT[ch][7] * inv);
    }
}

extern "C" void kernel_launch(void* const* d_in, const int* in_sizes, int n_in,
                              void* d_out, int out_size, void* d_ws, size_t ws_size,
                              hipStream_t stream) {
    const float* Q = (const float*)d_in[0];
    const float* K = (const float*)d_in[1];
    const float* V = (const float*)d_in[2];
    float* O = (float*)d_out;
    (void)in_sizes; (void)n_in; (void)out_size; (void)d_ws; (void)ws_size;

    dim3 grid(SEQ / 128, NHEADS);   // 8 q-blocks x 96 heads
    dim3 block(256);                // 8 waves x 16 query rows
    flash_attn_fwd_gfx1250<<<grid, block, 0, stream>>>(Q, K, V, O);
}
